// MiniBatchDiscrimination_79843442033274
// MI455X (gfx1250) — compile-verified, matrix operand order checked by reading, not run on hardware
//
#include <hip/hip_runtime.h>
#include <hip/hip_bf16.h>

// Problem constants (from reference)
#define NB   256      // batch
#define AF   2048     // in_features (GEMM K)
#define BF   64       // out_features
#define CF   32       // kernel dim
#define KD   2048     // = BF*CF = GEMM N dim
#define OSTR (AF + BF) // 2112 output row stride

typedef __attribute__((ext_vector_type(16))) _Float16 v16h;
typedef __attribute__((ext_vector_type(8)))  _Float16 v8h;
typedef __attribute__((ext_vector_type(8)))  float    v8f;

// ---------------------------------------------------------------------------
// Kernel 1: convert f (N x A, fp32 row-major) -> fp16 row-major
// ---------------------------------------------------------------------------
__global__ void __launch_bounds__(256) k_conv_f(const float* __restrict__ f,
                                                _Float16* __restrict__ fh) {
    int idx = blockIdx.x * blockDim.x + threadIdx.x;   // 0 .. N*A-1
    fh[idx] = (_Float16)f[idx];
}

// ---------------------------------------------------------------------------
// Kernel 2: convert + transpose T (K x N fp32 row-major) -> fp16 (N x K)
// so the WMMA B-fragment (column N, contiguous K) loads as b128s.
// ---------------------------------------------------------------------------
__global__ void __launch_bounds__(256) k_conv_T(const float* __restrict__ T,
                                                _Float16* __restrict__ Tt) {
    int idx = blockIdx.x * blockDim.x + threadIdx.x;   // 0 .. AF*KD-1
    int k = idx >> 11;          // / 2048  (row of T)
    int n = idx & (KD - 1);     // % 2048  (col of T)
    Tt[(size_t)n * AF + k] = (_Float16)T[idx];
}

// Load a 16-lane x 32-K fp16 WMMA fragment: per lane two contiguous 16B
// chunks (K {0..7 | 8..15} then K {16..23 | 24..31} selected by lane>=16).
__device__ __forceinline__ v16h load_frag(const _Float16* __restrict__ p) {
    v8h x0 = *(const v8h*)p;
    v8h x1 = *(const v8h*)(p + 16);
    v16h v;
#pragma unroll
    for (int e = 0; e < 8; ++e) { v[e] = x0[e]; v[8 + e] = x1[e]; }
    return v;
}

// ---------------------------------------------------------------------------
// Kernel 3: WMMA GEMM  M = f @ T   (256 x 2048) = (256 x 2048)(2048 x 2048)
// One wave per 32x64 output block: 2 A-frags x 4 B-frags -> 8 WMMAs per
// K=32 step off 12 b128 loads (2.7x less L2 traffic than 16x16 tiling).
// ---------------------------------------------------------------------------
__global__ void __launch_bounds__(32) k_gemm_wmma(const _Float16* __restrict__ Ah,
                                                  const _Float16* __restrict__ Bt,
                                                  float* __restrict__ Mo) {
    const int tileM = blockIdx.x * 32;   // 8 blocks of M rows
    const int tileN = blockIdx.y * 64;   // 32 blocks of M cols
    const int lane  = threadIdx.x;       // wave32
    const int l     = lane & 15;
    const int hi    = lane >> 4;         // 0 or 1

    const _Float16* ap = Ah + (size_t)(tileM + l) * AF + hi * 8;
    const _Float16* bp = Bt + (size_t)(tileN + l) * AF + hi * 8;
    const size_t rs16 = (size_t)16 * AF;   // 16-row stride in frags

    v8f acc[2][4] = {};
    for (int k = 0; k < AF; k += 32) {
        __builtin_prefetch(ap + k + 256, 0, 1);   // global_prefetch_b8
        __builtin_prefetch(bp + k + 256, 0, 1);
        v16h a[2], b[4];
#pragma unroll
        for (int m = 0; m < 2; ++m) a[m] = load_frag(ap + m * rs16 + k);
#pragma unroll
        for (int n = 0; n < 4; ++n) b[n] = load_frag(bp + n * rs16 + k);
#pragma unroll
        for (int m = 0; m < 2; ++m)
#pragma unroll
            for (int n = 0; n < 4; ++n)
                // (neg_a, A, neg_b, B, c_mod, C, reuse_a, reuse_b)
                acc[m][n] = __builtin_amdgcn_wmma_f32_16x16x32_f16(
                    false, a[m], false, b[n], (short)0, acc[m][n], false, false);
    }

    // C/D layout: VGPR r, lanes 0-15 -> (M=r, N=l); lanes 16-31 -> (M=r+8, N=l)
#pragma unroll
    for (int m = 0; m < 2; ++m) {
        float* mp = Mo + (size_t)(tileM + m * 16 + hi * 8) * KD + tileN + l;
#pragma unroll
        for (int n = 0; n < 4; ++n)
#pragma unroll
            for (int r = 0; r < 8; ++r)
                mp[n * 16 + (size_t)r * KD] = acc[m][n][r];
    }
}

// ---------------------------------------------------------------------------
// Kernel 4: pairwise L1 + exp reduction.
// One block per discrimination feature b; M[:, b, :] (256x32 f32 = 32 KB)
// staged in LDS; thread j keeps its own row in registers, sweeps i.
// o[j,b] = sum_i exp(-sum_c |M[i,b,c] - M[j,b,c]|)
// ---------------------------------------------------------------------------
__global__ void __launch_bounds__(256) k_pairwise(const float* __restrict__ Mo,
                                                  float* __restrict__ out) {
    __shared__ float sm[NB * CF];   // 32 KB
    const int b = blockIdx.x;       // 0..63
    const int j = threadIdx.x;      // 0..255

    for (int idx = threadIdx.x; idx < NB * CF; idx += 256) {
        int i = idx >> 5;           // row
        int c = idx & 31;           // kernel dim
        sm[idx] = Mo[(size_t)i * KD + b * CF + c];
    }
    __syncthreads();

    float mj[CF];
#pragma unroll
    for (int c = 0; c < CF; ++c) mj[c] = sm[j * CF + c];

    float acc = 0.0f;
    for (int i = 0; i < NB; ++i) {
        float l1 = 0.0f;
#pragma unroll
        for (int c = 0; c < CF; ++c) l1 += fabsf(sm[i * CF + c] - mj[c]);
        acc += __expf(-l1);
    }
    out[(size_t)j * OSTR + AF + b] = acc;
}

// ---------------------------------------------------------------------------
// Kernel 5: copy f into out[:, 0:2048] (float4 vectorized; 2048 % 4 == 0,
// 2112 % 4 == 0 so the strided output stays 16B aligned).
// ---------------------------------------------------------------------------
__global__ void __launch_bounds__(256) k_copy_f(const float4* __restrict__ f4,
                                                float4* __restrict__ out4) {
    int idx = blockIdx.x * blockDim.x + threadIdx.x;  // 0 .. N*A/4-1
    int n  = idx >> 9;          // / (2048/4)
    int a4 = idx & 511;         // % 512
    out4[(size_t)n * (OSTR / 4) + a4] = f4[idx];
}

// ---------------------------------------------------------------------------
extern "C" void kernel_launch(void* const* d_in, const int* in_sizes, int n_in,
                              void* d_out, int out_size, void* d_ws, size_t ws_size,
                              hipStream_t stream) {
    const float* f = (const float*)d_in[0];   // (256, 2048)
    const float* T = (const float*)d_in[1];   // (2048, 2048)
    float* out = (float*)d_out;               // (256, 2112)

    // Workspace layout (11 MB total):
    //   [0, 1MB)      f in fp16          (256*2048*2)
    //   [1MB, 9MB)    T^T in fp16        (2048*2048*2)
    //   [9MB, 11MB) M = f@T in fp32    (256*2048*4)
    char* ws = (char*)d_ws;
    _Float16* fh = (_Float16*)(ws);
    _Float16* Tt = (_Float16*)(ws + (size_t)(1u << 20));
    float*    Mo = (float*)   (ws + (size_t)(9u << 20));

    k_conv_f<<<(NB * AF) / 256, 256, 0, stream>>>(f, fh);
    k_conv_T<<<(AF * KD) / 256, 256, 0, stream>>>(T, Tt);
    k_gemm_wmma<<<dim3(NB / 32, KD / 64), 32, 0, stream>>>(fh, Tt, Mo);
    k_pairwise<<<BF, NB, 0, stream>>>(Mo, out);
    k_copy_f<<<(NB * AF / 4) / 256, 256, 0, stream>>>((const float4*)f, (float4*)out);
}
